// RCNN_74045236183269
// MI455X (gfx1250) — matrix operand order repaired and checked by reading, not compiled, not run on hardware
//
#include <hip/hip_runtime.h>

// ---------------------------------------------------------------------------
// CDNA5 (gfx1250) PointRCNN head. bf16 WMMA (v_wmma_f32_16x16x32_bf16) GEMMs,
// pre-packed fragment-swizzled weights, 32x64 output strips per wave, fused
// per-group SA MLPs with LDS-resident activations, and async global->LDS
// feature staging (global_load_async_to_lds_b128 + s_wait_asynccnt).
// Channel layout inside SA groups is features-first/xyz-last (weights are
// K-permuted to match) so async 16B copies are always aligned.
// ---------------------------------------------------------------------------

typedef __attribute__((ext_vector_type(16))) __bf16 bf16x16;
typedef __attribute__((ext_vector_type(8)))  __bf16 bf16x8;
typedef __attribute__((ext_vector_type(8)))  float  f32x8;

#define BB   64
#define NN   512
#define INCH 133   // 5 xyz-features + 128 rpn features

__device__ __forceinline__ __bf16 f2bf(float f) {
  unsigned u = __builtin_bit_cast(unsigned, f);
  unsigned r = u + 0x7FFFu + ((u >> 16) & 1u);   // round-to-nearest-even
  unsigned short h = (unsigned short)(r >> 16);
  return __builtin_bit_cast(__bf16, h);
}
__device__ __forceinline__ float bf2f(__bf16 x) {
  unsigned short h = __builtin_bit_cast(unsigned short, x);
  unsigned u = ((unsigned)h) << 16;
  return __builtin_bit_cast(float, u);
}

// A fragment: 16x32 bf16 tile (rows row0.., cols k0..k0+31) from row-major A.
// Per ISA 7.12.2 each lane holds two contiguous 8-element chunks -> 2x 16B loads.
__device__ __forceinline__ bf16x16 load_a_frag(const __bf16* A, int lda,
                                               int row0, int k0, int lane) {
  int m = lane & 15, h = lane >> 4;
  const __bf16* p = A + (size_t)(row0 + m) * lda + k0 + h * 8;
  bf16x8 c0 = *(const bf16x8*)p;         // K = k0 + h*8  .. +7
  bf16x8 c1 = *(const bf16x8*)(p + 16);  // K = k0+16+h*8 .. +7
  return __builtin_shufflevector(c0, c1, 0, 1, 2, 3, 4, 5, 6, 7,
                                 8, 9, 10, 11, 12, 13, 14, 15);
}

// B fragment from pre-swizzled weights: one aligned 32-byte vector load.
__device__ __forceinline__ bf16x16 load_b_packed(const __bf16* Wswz, int tile_o,
                                                 int nkb, int kb, int lane) {
  return ((const bf16x16*)Wswz)[(size_t)(tile_o * nkb + kb) * 32 + lane];
}

// D (16x16 f32 accumulator) -> bias + optional relu -> bf16 row-major store.
__device__ __forceinline__ void store_tile(const f32x8& c, __bf16* C, int ldc,
                                           int row0, int col0, const float* bias,
                                           int relu, int lane) {
  int nn = lane & 15, h = lane >> 4;
  float bv = bias[col0 + nn];
#pragma unroll
  for (int j = 0; j < 8; ++j) {
    float v = c[j] + bv;
    if (relu) v = fmaxf(v, 0.f);
    C[(size_t)(row0 + j + h * 8) * ldc + col0 + nn] = f2bf(v);
  }
}

// ---------------------------------------------------------------------------
// Pack one layer's fp32 W (O x K, row-major) into bf16 B-fragment layout,
// zero-padded to Kpad. If permC != 0, the K axis is permuted to the
// features-first/xyz-last layout used in the SA group LDS buffers:
//   k < permC        -> source channel k + 3        (features)
//   permC <= k < +3  -> source channel k - permC    (xyz)
//   else             -> zero pad
// ---------------------------------------------------------------------------
__global__ void pack_w_kernel(const float* __restrict__ W, int K, int Kpad, int O,
                              int permC, __bf16* __restrict__ out) {
  int i = blockIdx.x * blockDim.x + threadIdx.x;
  int nkb = Kpad >> 5;
  int total = (O >> 4) * nkb * 32 * 16;
  if (i >= total) return;
  int e    = i & 15;
  int lane = (i >> 4) & 31;
  int kb   = (i >> 9) % nkb;
  int to   = (i >> 9) / nkb;
  int k    = kb * 32 + (e < 8 ? (lane >> 4) * 8 + e : 16 + (lane >> 4) * 8 + (e - 8));
  int col  = to * 16 + (lane & 15);
  int ks = k;
  if (permC) ks = (k < permC) ? (k + 3) : ((k < permC + 3) ? (k - permC) : K);
  float x = (ks < K) ? W[(size_t)col * K + ks] : 0.f;
  out[i] = f2bf(x);
}

// ---------------------------------------------------------------------------
// GEMM: C[M x O] = relu(A[M x Kpad] * W^T + bias). A bf16 zero-padded to Kpad.
// One 32x64 strip per wave: each B fragment feeds 2 WMMAs.
// ---------------------------------------------------------------------------
__global__ void gemm_bias_relu_bf16(const __bf16* __restrict__ A, int lda,
                                    const __bf16* __restrict__ Wswz,
                                    const float* __restrict__ bias,
                                    __bf16* __restrict__ C, int ldc,
                                    int M, int Kpad, int O, int relu) {
  int lane = threadIdx.x & 31;
  int wave = blockIdx.x * (blockDim.x >> 5) + (threadIdx.x >> 5);
  int tmc = M >> 5, osc = O >> 6;
  if (wave >= tmc * osc) return;           // wave-uniform: EXEC stays all-ones
  int row0 = (wave % tmc) << 5;
  int os   = wave / tmc;
  int nkb  = Kpad >> 5;
  f32x8 acc0[4] = {}, acc1[4] = {};
  for (int kb = 0; kb < nkb; ++kb) {
    bf16x16 a0 = load_a_frag(A, lda, row0,      kb << 5, lane);
    bf16x16 a1 = load_a_frag(A, lda, row0 + 16, kb << 5, lane);
#pragma unroll
    for (int j = 0; j < 4; ++j) {
      bf16x16 b = load_b_packed(Wswz, os * 4 + j, nkb, kb, lane);
      acc0[j] = __builtin_amdgcn_wmma_f32_16x16x32_bf16(false, a0, false, b,
                                                        (short)0, acc0[j], false, false);
      acc1[j] = __builtin_amdgcn_wmma_f32_16x16x32_bf16(false, a1, false, b,
                                                        (short)0, acc1[j], false, false);
    }
  }
#pragma unroll
  for (int j = 0; j < 4; ++j) {
    store_tile(acc0[j], C, ldc, row0,      (os * 4 + j) << 4, bias, relu, lane);
    store_tile(acc1[j], C, ldc, row0 + 16, (os * 4 + j) << 4, bias, relu, lane);
  }
}

// One MLP layer computed LDS->LDS inside a 4-wave workgroup; 32x64 strips.
__device__ __forceinline__ void mlp_layer_lds(const __bf16* sin, int Kp,
                                              const __bf16* Wswz, const float* bias,
                                              __bf16* sout, int O, int nrows,
                                              int lane, int wave, int nwaves) {
  int tmc = nrows >> 5, osc = O >> 6, nkb = Kp >> 5;
  for (int s = wave; s < tmc * osc; s += nwaves) {   // wave-uniform loop
    int row0 = (s % tmc) << 5;
    int os   = s / tmc;
    f32x8 acc0[4] = {}, acc1[4] = {};
    for (int kb = 0; kb < nkb; ++kb) {
      bf16x16 a0 = load_a_frag(sin, Kp, row0,      kb << 5, lane);
      bf16x16 a1 = load_a_frag(sin, Kp, row0 + 16, kb << 5, lane);
#pragma unroll
      for (int j = 0; j < 4; ++j) {
        bf16x16 b = load_b_packed(Wswz, os * 4 + j, nkb, kb, lane);
        acc0[j] = __builtin_amdgcn_wmma_f32_16x16x32_bf16(false, a0, false, b,
                                                          (short)0, acc0[j], false, false);
        acc1[j] = __builtin_amdgcn_wmma_f32_16x16x32_bf16(false, a1, false, b,
                                                          (short)0, acc1[j], false, false);
      }
    }
#pragma unroll
    for (int j = 0; j < 4; ++j) {
      store_tile(acc0[j], sout, O, row0,      (os * 4 + j) << 4, bias, 1, lane);
      store_tile(acc1[j], sout, O, row0 + 16, (os * 4 + j) << 4, bias, 1, lane);
    }
  }
}

// ---------------------------------------------------------------------------
// Fused set-abstraction kernel: one workgroup (4 waves) per group.
// LDS row layout (pitch Kpad0): [feat 0..featC-1 | xyz 3ch | zero pad].
// Feature blocks are staged with async global->LDS 16B copies (ASYNCcnt),
// xyz/pad written with normal stores; s_wait_asynccnt + barrier before WMMA.
// gidx==nullptr => identity rows; centers==nullptr => absolute xyz (SA2).
// ---------------------------------------------------------------------------
__global__ void sa_fused_kernel(const float* __restrict__ xyz,
                                const float* __restrict__ centers,
                                const __bf16* __restrict__ feat, int n, int featC,
                                const int* __restrict__ gidx, int npoint, int nrows,
                                int Kpad0,
                                const __bf16* __restrict__ W0, const float* __restrict__ B0, int O0,
                                const __bf16* __restrict__ W1, const float* __restrict__ B1, int O1,
                                const __bf16* __restrict__ W2, const float* __restrict__ B2, int O2,
                                int in_cap,
                                __bf16* __restrict__ outfeat) {
  extern __shared__ __bf16 smem[];
  __bf16* s_in  = smem;           // capacity in_cap elements
  __bf16* s_out = smem + in_cap;  // capacity max(O0,O2)*nrows elements

  int g = blockIdx.x;
  int b = g / npoint;
  int tid = threadIdx.x;
  int lane = tid & 31, wave = tid >> 5, nwaves = blockDim.x >> 5;

  const int*    gi    = gidx ? gidx + (size_t)g * nrows : nullptr;
  const float*  ctr   = centers ? centers + (size_t)g * 3 : nullptr;
  const float*  bxyz  = xyz  + (size_t)b * n * 3;
  const __bf16* bfeat = feat + (size_t)b * n * featC;

  // --- async-stage the per-row feature blocks into LDS (16B per lane) ---
  unsigned lds_base = __builtin_amdgcn_groupstaticsize();  // dynamic-LDS offset
  int cshift = (featC == 128) ? 4 : 5;       // 16B chunks per row = featC/8
  int chunks = nrows << cshift;              // multiples of 128: no tail
  for (int ch = tid; ch < chunks; ch += blockDim.x) {
    int r = ch >> cshift;
    int part = ch & ((1 << cshift) - 1);
    int j = gi ? gi[r] : r;
    unsigned long long ga =
        (unsigned long long)(const void*)(bfeat + (size_t)j * featC + part * 8);
    unsigned la = lds_base + (unsigned)((r * Kpad0 + part * 8) * 2);
    asm volatile("global_load_async_to_lds_b128 %0, %1, off"
                 :: "v"(la), "v"(ga) : "memory");
  }
  // --- xyz (relative) channels + zero pad, normal stores ---
  {
    int w = Kpad0 - featC;                   // 32 trailing channels
    for (int t = tid; t < nrows * w; t += blockDim.x) {
      int r = t / w, c0 = t % w;
      float v = 0.f;
      if (c0 < 3) {
        int j = gi ? gi[r] : r;
        v = bxyz[j * 3 + c0] - (ctr ? ctr[c0] : 0.f);
      }
      s_in[r * Kpad0 + featC + c0] = f2bf(v);
    }
  }
  asm volatile("s_wait_asynccnt 0x0" ::: "memory");
  __syncthreads();

  mlp_layer_lds(s_in,  Kpad0, W0, B0, s_out, O0, nrows, lane, wave, nwaves);
  __syncthreads();
  mlp_layer_lds(s_out, O0,    W1, B1, s_in,  O1, nrows, lane, wave, nwaves);
  __syncthreads();
  mlp_layer_lds(s_in,  O1,    W2, B2, s_out, O2, nrows, lane, wave, nwaves);
  __syncthreads();

  // channel-wise max-pool over the nrows samples
  for (int c = tid; c < O2; c += blockDim.x) {
    float m = -3.4e38f;
    for (int r = 0; r < nrows; ++r) m = fmaxf(m, bf2f(s_out[r * O2 + c]));
    outfeat[(size_t)g * O2 + c] = f2bf(m);
  }
}

// ---------------------------------------------------------------------------
// Farthest-point sampling: one block per batch, sequential scan with LDS
// points/distances and an argmax tree reduction (matches the lax.scan).
// ---------------------------------------------------------------------------
__global__ void fps_kernel(const float* __restrict__ xyz, int n, int npoint,
                           int* __restrict__ idx_out, float* __restrict__ newxyz) {
  __shared__ float sx[512], sy[512], sz[512], sd[512];
  __shared__ float rmax[256];
  __shared__ int   rarg[256];
  int b = blockIdx.x, tid = threadIdx.x;
  const float* p = xyz + (size_t)b * n * 3;
  for (int i = tid; i < n; i += blockDim.x) {
    sx[i] = p[i * 3]; sy[i] = p[i * 3 + 1]; sz[i] = p[i * 3 + 2];
    sd[i] = 1e10f;
  }
  __syncthreads();
  int far = 0;
  for (int it = 0; it < npoint; ++it) {
    if (tid == 0) {
      idx_out[b * npoint + it] = far;
      newxyz[(size_t)(b * npoint + it) * 3 + 0] = sx[far];
      newxyz[(size_t)(b * npoint + it) * 3 + 1] = sy[far];
      newxyz[(size_t)(b * npoint + it) * 3 + 2] = sz[far];
    }
    float cx = sx[far], cy = sy[far], cz = sz[far];
    float best = -1.f; int bestj = 0;
    for (int j = tid; j < n; j += blockDim.x) {
      float dx = sx[j] - cx, dy = sy[j] - cy, dz = sz[j] - cz;
      float nd = fminf(sd[j], dx * dx + dy * dy + dz * dz);
      sd[j] = nd;
      if (nd > best) { best = nd; bestj = j; }
    }
    rmax[tid] = best; rarg[tid] = bestj;
    __syncthreads();
    for (int s = blockDim.x >> 1; s > 0; s >>= 1) {
      if (tid < s) {
        float o = rmax[tid + s]; int oj = rarg[tid + s];
        if (o > rmax[tid] || (o == rmax[tid] && oj < rarg[tid])) {
          rmax[tid] = o; rarg[tid] = oj;
        }
      }
      __syncthreads();
    }
    far = rarg[0];
    __syncthreads();
  }
}

// Ball query: ascending scan keeps the same indices as sort()[:nsample].
__global__ void ballq_kernel(const float* __restrict__ xyz,
                             const float* __restrict__ newxyz, int n, int npoint,
                             float r2, int nsample, int total,
                             int* __restrict__ out) {
  int g = blockIdx.x * blockDim.x + threadIdx.x;
  if (g >= total) return;
  int b = g / npoint;
  const float* p = xyz + (size_t)b * n * 3;
  const float* q = newxyz + (size_t)g * 3;
  int* o = out + (size_t)g * nsample;
  int cnt = 0;
  for (int j = 0; j < n && cnt < nsample; ++j) {
    float dx = p[j * 3] - q[0], dy = p[j * 3 + 1] - q[1], dz = p[j * 3 + 2] - q[2];
    if (dx * dx + dy * dy + dz * dz <= r2) o[cnt++] = j;
  }
  if (cnt == 0) { for (int i = 0; i < nsample; ++i) o[i] = n - 1; }
  else          { int f0 = o[0]; for (int i = cnt; i < nsample; ++i) o[i] = f0; }
}

// Split pts_input into contiguous xyz(f32), padded 5-ch bf16 A, rpn bf16.
__global__ void prep_kernel(const float* __restrict__ pts, int BN,
                            float* __restrict__ xyz, __bf16* __restrict__ axyz,
                            __bf16* __restrict__ rpn) {
  int i = blockIdx.x * blockDim.x + threadIdx.x;
  if (i >= BN) return;
  const float* s = pts + (size_t)i * INCH;
  xyz[(size_t)i * 3 + 0] = s[0];
  xyz[(size_t)i * 3 + 1] = s[1];
  xyz[(size_t)i * 3 + 2] = s[2];
  for (int c = 0; c < 32; ++c)  axyz[(size_t)i * 32 + c]  = f2bf(c < 5 ? s[c] : 0.f);
  for (int c = 0; c < 128; ++c) rpn[(size_t)i * 128 + c]  = f2bf(s[5 + c]);
}

__global__ void concat_bf16(const __bf16* __restrict__ a, const __bf16* __restrict__ b,
                            __bf16* __restrict__ o, int rows, int ca, int cb) {
  int i = blockIdx.x * blockDim.x + threadIdx.x;
  int cw = ca + cb;
  if (i >= rows * cw) return;
  int r = i / cw, c = i % cw;
  o[i] = (c < ca) ? a[(size_t)r * ca + c] : b[(size_t)r * cb + (c - ca)];
}

// Final head layer (output width not a multiple of 16): plain dot products.
__global__ void head_final(const __bf16* __restrict__ h, int K,
                           const float* __restrict__ W, const float* __restrict__ bias,
                           float* __restrict__ out, int Bn, int O) {
  int i = blockIdx.x * blockDim.x + threadIdx.x;
  if (i >= Bn * O) return;
  int b = i / O, o = i % O;
  float acc = bias[o];
  const __bf16* hp = h + (size_t)b * K;
  const float*  wp = W + (size_t)o * K;
  for (int k = 0; k < K; ++k) acc += bf2f(hp[k]) * wp[k];
  out[i] = acc;
}

// ---------------------------------------------------------------------------
extern "C" void kernel_launch(void* const* d_in, const int* in_sizes, int n_in,
                              void* d_out, int out_size, void* d_ws, size_t ws_size,
                              hipStream_t stream) {
  (void)n_in; (void)out_size; (void)ws_size;

  // Logical layer order: xyz_up(2), merge(1), sa0(3), sa1(3), sa2(3),
  //                      cls(3), reg(3), iou(3), link(3), se(3)  -> 27 layers.
  static const int MAP_INS[27] = { 1,3, 5, 7,9,11, 13,15,17, 19,21,23,
                                   25,27,29, 31,33,35, 37,39,41, 43,45,47, 49,51,53 };
  static const int MAP_SRT[27] = { 51,53, 19, 27,29,31, 33,35,37, 39,41,43,
                                   1,3,5, 21,23,25, 7,9,11, 13,15,17, 45,47,49 };
  const int* MP = (in_sizes[1] == 640) ? MAP_INS : MAP_SRT;  // xyz_up W0 is 128x5
  auto Wp = [&](int l) { return (const float*)d_in[MP[l]]; };
  auto Bp = [&](int l) { return (const float*)d_in[MP[l] + 1]; };

  const float* pts = (const float*)d_in[0];
  const int BN = BB * NN;  // 32768 points

  // workspace bump allocator
  char* wp_ = (char*)d_ws;
  auto alloc = [&](size_t bytes) -> void* {
    void* p = (void*)wp_; wp_ += (bytes + 255) & ~(size_t)255; return p;
  };
  float*  xyzf    = (float*) alloc((size_t)BN * 3 * 4);
  __bf16* axyz    = (__bf16*)alloc((size_t)BN * 32 * 2);
  __bf16* rpn     = (__bf16*)alloc((size_t)BN * 128 * 2);
  __bf16* act1    = (__bf16*)alloc((size_t)BN * 128 * 2);
  __bf16* xyzfeat = (__bf16*)alloc((size_t)BN * 128 * 2);
  __bf16* merged  = (__bf16*)alloc((size_t)BN * 256 * 2);
  __bf16* feat0   = (__bf16*)alloc((size_t)BN * 128 * 2);
  int*    fidx0   = (int*)   alloc((size_t)BB * 128 * 4);
  float*  nxyz0   = (float*) alloc((size_t)BB * 128 * 3 * 4);
  int*    gidx0   = (int*)   alloc((size_t)BB * 128 * 64 * 4);
  __bf16* feat1   = (__bf16*)alloc((size_t)BB * 128 * 128 * 2);
  int*    fidx1   = (int*)   alloc((size_t)BB * 32 * 4);
  float*  nxyz1   = (float*) alloc((size_t)BB * 32 * 3 * 4);
  int*    gidx1   = (int*)   alloc((size_t)BB * 32 * 64 * 4);
  __bf16* feat2   = (__bf16*)alloc((size_t)BB * 32 * 256 * 2);
  __bf16* fvec    = (__bf16*)alloc((size_t)BB * 512 * 2);
  __bf16* h1      = (__bf16*)alloc((size_t)BB * 256 * 2);
  __bf16* h2      = (__bf16*)alloc((size_t)BB * 256 * 2);

  // ---- pack all WMMA-consumed weights: bf16, padded, fragment-swizzled ----
  // perm = featC for SA layer-1 weights (features-first/xyz-last LDS layout).
  struct LP { int l, K, Kpad, O, perm; };
  static const LP LPT[] = {
    {0,5,32,128,0},      {1,128,128,128,0},  {2,256,256,128,0},
    {3,131,160,128,128}, {4,128,128,128,0},  {5,128,128,128,0},
    {6,131,160,128,128}, {7,128,128,128,0},  {8,128,128,256,0},
    {9,259,288,256,256}, {10,256,256,256,0}, {11,256,256,512,0},
    {12,512,512,256,0},  {13,256,256,256,0},
    {15,512,512,256,0},  {16,256,256,256,0},
    {18,512,512,256,0},  {19,256,256,256,0},
    {21,512,512,256,0},  {22,256,256,256,0},
    {24,512,512,256,0},  {25,256,256,256,0},
  };
  __bf16* packed[27] = {};
  for (int i = 0; i < (int)(sizeof(LPT) / sizeof(LPT[0])); ++i) {
    const LP& e = LPT[i];
    int total = (e.O >> 4) * (e.Kpad >> 5) * 512;
    packed[e.l] = (__bf16*)alloc((size_t)total * 2);
    pack_w_kernel<<<(total + 255) / 256, 256, 0, stream>>>(Wp(e.l), e.K, e.Kpad,
                                                           e.O, e.perm, packed[e.l]);
  }

  auto gemm = [&](const __bf16* A, int lda, int l, __bf16* C, int ldc,
                  int M, int Kpad, int O) {
    int waves = (M >> 5) * (O >> 6);
    gemm_bias_relu_bf16<<<(waves + 3) / 4, 128, 0, stream>>>(
        A, lda, packed[l], Bp(l), C, ldc, M, Kpad, O, 1);
  };

  // ---- point-wise MLPs -------------------------------------------------
  prep_kernel<<<(BN + 255) / 256, 256, 0, stream>>>(pts, BN, xyzf, axyz, rpn);
  gemm(axyz, 32, 0, act1, 128, BN, 32, 128);        // xyz_up L0: 5 -> 128
  gemm(act1, 128, 1, xyzfeat, 128, BN, 128, 128);   // xyz_up L1
  {
    int tot = BN * 256;
    concat_bf16<<<(tot + 255) / 256, 256, 0, stream>>>(xyzfeat, rpn, merged, BN, 128, 128);
  }
  gemm(merged, 256, 2, feat0, 128, BN, 256, 128);   // merge: 256 -> 128

  // ---- SA stage 0: 512 pts -> 128 centroids ---------------------------
  fps_kernel<<<BB, 256, 0, stream>>>(xyzf, NN, 128, fidx0, nxyz0);
  ballq_kernel<<<(BB * 128 + 255) / 256, 256, 0, stream>>>(xyzf, nxyz0, NN, 128,
                                                           0.04f, 64, BB * 128, gidx0);
  {
    int in_cap = 64 * 160, out_cap = 64 * 128;
    size_t smem = (size_t)(in_cap + out_cap) * 2;
    sa_fused_kernel<<<BB * 128, 128, smem, stream>>>(
        xyzf, nxyz0, feat0, NN, 128, gidx0, 128, 64, 160,
        packed[3], Bp(3), 128, packed[4], Bp(4), 128, packed[5], Bp(5), 128,
        in_cap, feat1);
  }

  // ---- SA stage 1: 128 pts -> 32 centroids ----------------------------
  fps_kernel<<<BB, 256, 0, stream>>>(nxyz0, 128, 32, fidx1, nxyz1);
  ballq_kernel<<<(BB * 32 + 255) / 256, 256, 0, stream>>>(nxyz0, nxyz1, 128, 32,
                                                          0.16f, 64, BB * 32, gidx1);
  {
    int in_cap = 64 * 160, out_cap = 64 * 256;
    size_t smem = (size_t)(in_cap + out_cap) * 2;
    sa_fused_kernel<<<BB * 32, 128, smem, stream>>>(
        nxyz0, nxyz1, feat1, 128, 128, gidx1, 32, 64, 160,
        packed[6], Bp(6), 128, packed[7], Bp(7), 128, packed[8], Bp(8), 256,
        in_cap, feat2);
  }

  // ---- SA stage 2 (global): 32 rows of concat(feat, xyz) --------------
  {
    int in_cap = 32 * 288, out_cap = 32 * 512;
    size_t smem = (size_t)(in_cap + out_cap) * 2;
    sa_fused_kernel<<<BB, 128, smem, stream>>>(
        nxyz1, nullptr, feat2, 32, 256, nullptr, 1, 32, 288,
        packed[9], Bp(9), 256, packed[10], Bp(10), 256, packed[11], Bp(11), 512,
        in_cap, fvec);
  }

  // ---- heads ----------------------------------------------------------
  static const int HEADL[5]  = { 12, 15, 18, 21, 24 };   // cls, reg, iou, link, se
  static const int OUTDIM[5] = { 1, 46, 1, 1, 1 };
  static const int OUTOFF[5] = { 0, 64, 3008, 3072, 3136 };
  for (int hix = 0; hix < 5; ++hix) {
    int l = HEADL[hix];
    gemm(fvec, 512, l,     h1, 256, BB, 512, 256);
    gemm(h1,   256, l + 1, h2, 256, BB, 256, 256);
    int tot = BB * OUTDIM[hix];
    head_final<<<(tot + 255) / 256, 256, 0, stream>>>(
        h2, 256, Wp(l + 2), Bp(l + 2), (float*)d_out + OUTOFF[hix], BB, OUTDIM[hix]);
  }
}
